// GeometricTransformer_66090956751034
// MI455X (gfx1250) — matrix operand order repaired and checked by reading, not compiled, not run on hardware
//
#include <hip/hip_runtime.h>
#include <hip/hip_bf16.h>
#include <math.h>

#define NPTS 320
#define NN (NPTS*NPTS)
#define HD 256
#define NHEADS 4
#define CHEAD 64
#define CIN 1024
#define CDIV(a,b) (((a)+(b)-1)/(b))

typedef __attribute__((ext_vector_type(16))) _Float16 v16h;
typedef __attribute__((ext_vector_type(8)))  _Float16 v8h;
typedef __attribute__((ext_vector_type(8)))  float    v8f;

// ---------------- WMMA helpers (gfx1250 wave32, 16x16x32 f16 -> f32) -------
static __device__ __forceinline__ v8f wmma16(v16h a, v16h b, v8f c) {
  return __builtin_amdgcn_wmma_f32_16x16x32_f16(false, a, false, b, (short)0, c,
                                                false, false);
}
// A fragment: row = rowBase + (lane&15); k = h*16 + e (contiguous 16 halfs),
// h = lane>>4. Two 16B loads + concat.
static __device__ __forceinline__ v16h afrag(const _Float16* base, int rowBase,
                                             int lda, int kbase, int lane) {
  const _Float16* p = base + (size_t)(rowBase + (lane & 15)) * lda + kbase +
                      ((lane >> 4) << 4);
  v8h lo = *(const v8h*)p;
  v8h hi = *(const v8h*)(p + 8);
  return __builtin_shufflevector(lo, hi, 0,1,2,3,4,5,6,7,8,9,10,11,12,13,14,15);
}
// B fragment: col = colBase + (lane&15); k = h*16 + e (contiguous 16 halfs).
// Source is row-major [col][k] (i.e. transposed weights, or k-rows / vT-rows).
static __device__ __forceinline__ v16h bfrag(const _Float16* base, int colBase,
                                             int lda, int kbase, int lane) {
  const _Float16* p = base + (size_t)(colBase + (lane & 15)) * lda + kbase +
                      ((lane >> 4) << 4);
  return *(const v16h*)p;
}
static __device__ __forceinline__ float wsum(float v) {
#pragma unroll
  for (int m = 16; m >= 1; m >>= 1) v += __shfl_xor(v, m, 32);
  return v;
}
static __device__ __forceinline__ float wmaxr(float v) {
#pragma unroll
  for (int m = 16; m >= 1; m >>= 1) v = fmaxf(v, __shfl_xor(v, m, 32));
  return v;
}

// ---------------- weight transpose + f16 convert: out[n*K+k] = W[k*Nc+n] ----
__global__ void k_packT(const float* __restrict__ W, _Float16* __restrict__ out,
                        int K, int Nc) {
  int gid = blockIdx.x * blockDim.x + threadIdx.x;
  if (gid >= K * Nc) return;
  int n = gid / K, k = gid % K;
  out[gid] = (_Float16)W[(size_t)k * Nc + n];
}

__global__ void k_f2h(const float* __restrict__ x, _Float16* __restrict__ y, int n) {
  int i = blockIdx.x * blockDim.x + threadIdx.x;
  if (i < n) y[i] = (_Float16)x[i];
}

// ---------------- geometry scalar kernels ----------------------------------
__global__ void k_pairdist(const float* __restrict__ p0, const float* __restrict__ p1,
                           float* __restrict__ dist) {
  int gid = blockIdx.x * blockDim.x + threadIdx.x;
  int cloud = blockIdx.y;
  if (gid >= NN) return;
  const float* P = cloud ? p1 : p0;
  int n = gid / NPTS, m = gid % NPTS;
  float dx = P[n*3]-P[m*3], dy = P[n*3+1]-P[m*3+1], dz = P[n*3+2]-P[m*3+2];
  dist[(size_t)cloud*NN + gid] = sqrtf(fmaxf(dx*dx + dy*dy + dz*dz, 1e-12f));
}

// one thread per (cloud,row): 26 smallest distances -> knn idx + entropies
__global__ void k_knn(const float* __restrict__ dist, int* __restrict__ knn,
                      float* __restrict__ ent) {
  int t = blockIdx.x * blockDim.x + threadIdx.x;
  if (t >= 2 * NPTS) return;
  int cloud = t / NPTS, n = t % NPTS;
  const float* row = dist + (size_t)cloud*NN + (size_t)n*NPTS;
  float best[26]; int bidx[26];
  for (int i = 0; i < 26; i++) { best[i] = 3.4e38f; bidx[i] = 0; }
  for (int j = 0; j < NPTS; j++) {
    float d = row[j];
    if (d < best[25]) {
      int p = 25;
      while (p > 0 && best[p-1] > d) { best[p] = best[p-1]; bidx[p] = bidx[p-1]; p--; }
      best[p] = d; bidx[p] = j;
    }
  }
  for (int k = 0; k < 3; k++) knn[(cloud*NPTS + n)*3 + k] = bidx[1 + k];
  const int kv[3] = {5, 10, 25};
  for (int e = 0; e < 3; e++) {
    int kk = kv[e];
    float s = 0.f; for (int i = 1; i <= kk; i++) s += best[i];
    float en = 0.f;
    for (int i = 1; i <= kk; i++) { float pp = best[i] / s; en -= pp * logf(pp + 1e-10f); }
    ent[(cloud*NPTS + n)*3 + e] = en;
  }
}

__global__ void k_density(const float* __restrict__ dist, float* __restrict__ den) {
  int t = blockIdx.x * blockDim.x + threadIdx.x;
  if (t >= 2 * NPTS) return;
  int cloud = t / NPTS, n = t % NPTS;
  const float* row = dist + (size_t)cloud*NN + (size_t)n*NPTS;
  float c0 = 0.f, c1 = 0.f, c2 = 0.f;
  for (int j = 0; j < NPTS; j++) {
    float d = row[j];
    c0 += (d < 0.5f) ? 1.f : 0.f; c1 += (d < 1.0f) ? 1.f : 0.f; c2 += (d < 2.0f) ? 1.f : 0.f;
  }
  den[t*3+0] = c0 / 0.52359878f;
  den[t*3+1] = c1 / 4.18879020f;
  den[t*3+2] = c2 / 33.5103216f;
}

__global__ void k_gradient(const float* __restrict__ den, float* __restrict__ grad) {
  int t = blockIdx.x * blockDim.x + threadIdx.x;
  if (t >= 2 * NPTS * 3) return;
  int cloud = t / (NPTS*3); int rem = t % (NPTS*3); int n = rem / 3, s = rem % 3;
  float dn = den[(cloud*NPTS + n)*3 + s];
  float acc = 0.f;
  for (int m = 0; m < NPTS; m++) { float d = dn - den[(cloud*NPTS + m)*3 + s]; acc += d*d; }
  grad[t] = sqrtf(fmaxf(acc, 1e-12f));
}

// per (cloud,n): e2[n][c] = normalize(mdve@mW+mb) + normalize(lgee@lW+lb)
__global__ void k_neighemb(const float* __restrict__ den, const float* __restrict__ grad,
                           const float* __restrict__ ent,
                           const float* __restrict__ mW, const float* __restrict__ mb,
                           const float* __restrict__ lW, const float* __restrict__ lb,
                           float* __restrict__ e2) {
  __shared__ float red[256];
  int bid = blockIdx.x; int cloud = bid / NPTS, n = bid % NPTS;
  int c = threadIdx.x;
  float v6[6], v3[3];
  for (int i = 0; i < 3; i++) v6[i]   = den[(cloud*NPTS + n)*3 + i];
  for (int i = 0; i < 3; i++) v6[3+i] = grad[(cloud*NPTS + n)*3 + i];
  float nm = 0.f; for (int i = 0; i < 6; i++) nm += v6[i]*v6[i];
  nm = fmaxf(sqrtf(nm), 1e-12f);
  for (int i = 0; i < 6; i++) v6[i] /= nm;
  for (int i = 0; i < 3; i++) v3[i] = ent[(cloud*NPTS + n)*3 + i];
  float nl = 0.f; for (int i = 0; i < 3; i++) nl += v3[i]*v3[i];
  nl = fmaxf(sqrtf(nl), 1e-12f);
  for (int i = 0; i < 3; i++) v3[i] /= nl;
  float mp = mb[c]; for (int i = 0; i < 6; i++) mp += v6[i]*mW[i*HD + c];
  float lp = lb[c]; for (int i = 0; i < 3; i++) lp += v3[i]*lW[i*HD + c];
  red[c] = mp*mp; __syncthreads();
  for (int s = 128; s > 0; s >>= 1) { if (c < s) red[c] += red[c+s]; __syncthreads(); }
  float nmm = fmaxf(sqrtf(red[0]), 1e-12f); __syncthreads();
  red[c] = lp*lp; __syncthreads();
  for (int s = 128; s > 0; s >>= 1) { if (c < s) red[c] += red[c+s]; __syncthreads(); }
  float nll = fmaxf(sqrtf(red[0]), 1e-12f);
  e2[((size_t)cloud*NPTS + n)*HD + c] = mp / nmm + lp / nll;
}

__global__ void k_aidx(const float* __restrict__ p0, const float* __restrict__ p1,
                       const int* __restrict__ knn, float* __restrict__ aidx) {
  int gid = blockIdx.x * blockDim.x + threadIdx.x;
  int cloud = blockIdx.y;
  if (gid >= NN) return;
  const float* P = cloud ? p1 : p0;
  int n = gid / NPTS, m = gid % NPTS;
  float pnx = P[n*3], pny = P[n*3+1], pnz = P[n*3+2];
  float ax = P[m*3]-pnx, ay = P[m*3+1]-pny, az = P[m*3+2]-pnz;
  for (int k = 0; k < 3; k++) {
    int j = knn[(cloud*NPTS + n)*3 + k];
    float rx = P[j*3]-pnx, ry = P[j*3+1]-pny, rz = P[j*3+2]-pnz;
    float cx = ry*az - rz*ay, cy = rz*ax - rx*az, cz = rx*ay - ry*ax;
    float sv = sqrtf(cx*cx + cy*cy + cz*cz);
    float cv = rx*ax + ry*ay + rz*az;
    aidx[((size_t)cloud*NN + gid)*3 + k] = atan2f(sv, cv) * 3.81971863f; // 180/(15*pi)
  }
}

// ---------------- geometric-embedding WMMA kernel --------------------------
// WG = 16 pairs; LDS-stage sinusoidal A rows for d + 3 angle parts, then each
// wave does 4 column-tiles x 8 K-steps x 4 WMMAs with max-fused epilogue.
__global__ void __launch_bounds__(128) k_geo(const float* __restrict__ dist,
    const float* __restrict__ aidx, const _Float16* __restrict__ WdT,
    const _Float16* __restrict__ WaT, const float* __restrict__ bd,
    const float* __restrict__ ba, _Float16* __restrict__ emb) {
  __shared__ _Float16 As[4 * 16 * HD];                 // 32 KB
  int pt = blockIdx.x, cloud = blockIdx.y, tid = threadIdx.x;
  for (int e = tid; e < 4*16*HD; e += 128) {
    int part = e >> 12, r = (e >> 8) & 15, kk = e & 255;
    int pair = pt*16 + r;
    float val = (part == 0) ? dist[(size_t)cloud*NN + pair] * 5.0f
                            : aidx[((size_t)cloud*NN + pair)*3 + (part-1)];
    float ang = val * __expf(-0.071955784f * (float)(kk >> 1));
    As[e] = (_Float16)((kk & 1) ? __cosf(ang) : __sinf(ang));
  }
  __syncthreads();
  int wave = tid >> 5, lane = tid & 31;
  for (int ci = 0; ci < 4; ci++) {
    int ct = wave*4 + ci;
    v8f ad = {0,0,0,0,0,0,0,0}, a0 = ad, a1 = ad, a2 = ad;
    for (int ks = 0; ks < 8; ks++) {
      int kb = ks*32;
      v16h bdw = bfrag(WdT, ct*16, HD, kb, lane);
      v16h baw = bfrag(WaT, ct*16, HD, kb, lane);
      v16h fd = afrag(As + 0*16*HD, 0, HD, kb, lane);
      v16h f0 = afrag(As + 1*16*HD, 0, HD, kb, lane);
      v16h f1 = afrag(As + 2*16*HD, 0, HD, kb, lane);
      v16h f2 = afrag(As + 3*16*HD, 0, HD, kb, lane);
      ad = wmma16(fd, bdw, ad);
      a0 = wmma16(f0, baw, a0);
      a1 = wmma16(f1, baw, a1);
      a2 = wmma16(f2, baw, a2);
    }
    int col = ct*16 + (lane & 15);
    float bb = bd[col] + ba[col];
#pragma unroll
    for (int r = 0; r < 8; r++) {
      int pair = pt*16 + ((lane >> 4) << 3) + r;
      float am = fmaxf(a0[r], fmaxf(a1[r], a2[r]));
      emb[((size_t)cloud*NN + pair)*HD + col] = (_Float16)(ad[r] + am + bb);
    }
  }
}

// emb = g*e1 + (1-g)*e2,   g = sigmoid(e1.selW + selb); wave per pair
__global__ void k_gate(_Float16* __restrict__ emb, const float* __restrict__ e2,
                       const float* __restrict__ selW, const float* __restrict__ selb) {
  int gw = blockIdx.x * (blockDim.x >> 5) + (threadIdx.x >> 5);
  int lane = threadIdx.x & 31;
  if (gw >= 2 * NN) return;
  int cloud = gw / NN, pair = gw % NN, m = pair % NPTS;
  _Float16* e = emb + ((size_t)cloud*NN + pair)*HD;
  const float* e2r = e2 + ((size_t)cloud*NPTS + m)*HD;
  float ev[8], s = 0.f;
#pragma unroll
  for (int i = 0; i < 8; i++) { float v = (float)e[lane*8 + i]; ev[i] = v; s += v * selW[lane*8 + i]; }
  s = wsum(s);
  float g = 1.0f / (1.0f + __expf(-(s + selb[0])));
#pragma unroll
  for (int i = 0; i < 8; i++)
    e[lane*8 + i] = (_Float16)(g * ev[i] + (1.0f - g) * e2r[lane*8 + i]);
}

// ---------------- generic WMMA GEMM: A(MxK f16) * WT(NcxK f16) + bias ------
__global__ void __launch_bounds__(128) k_gemm(const _Float16* __restrict__ A,
    const _Float16* __restrict__ BT, const float* __restrict__ bias,
    float* __restrict__ outF, _Float16* __restrict__ outH,
    _Float16* __restrict__ outHT, int M, int K, int Nc, int relu) {
  int gw = blockIdx.x * (blockDim.x >> 5) + (threadIdx.x >> 5);
  int lane = threadIdx.x & 31;
  int tilesM = M >> 4;
  if (gw >= tilesM * (Nc >> 4)) return;
  int tm = gw % tilesM, tn = gw / tilesM;
  v8f acc = {0,0,0,0,0,0,0,0};
  int ksteps = K >> 5;
  for (int ks = 0; ks < ksteps; ks++) {
    v16h a = afrag(A, tm*16, K, ks*32, lane);
    v16h b = bfrag(BT, tn*16, K, ks*32, lane);
    acc = wmma16(a, b, acc);
  }
  int n = tn*16 + (lane & 15);
  float bv = bias ? bias[n] : 0.0f;
#pragma unroll
  for (int r = 0; r < 8; r++) {
    int m = tm*16 + ((lane >> 4) << 3) + r;
    float v = acc[r] + bv;
    if (relu) v = fmaxf(v, 0.0f);
    if (outF)  outF[(size_t)m*Nc + n] = v;
    if (outH)  outH[(size_t)m*Nc + n] = (_Float16)v;
    if (outHT) outHT[(size_t)n*M + m] = (_Float16)v;
  }
}

// scores[h][n][m] = q[n,h*64:] . k[m,h*64:]
__global__ void __launch_bounds__(128) k_scores(const _Float16* __restrict__ qh,
    const _Float16* __restrict__ kh, float* __restrict__ sc) {
  int gw = blockIdx.x * 4 + (threadIdx.x >> 5);
  int lane = threadIdx.x & 31;
  const int TM = NPTS / 16;
  if (gw >= NHEADS * TM * TM) return;
  int h = gw / (TM*TM); int rem = gw % (TM*TM);
  int tm = rem % TM, tn = rem / TM;
  v8f acc = {0,0,0,0,0,0,0,0};
  for (int ks = 0; ks < 2; ks++) {
    v16h a = afrag(qh, tm*16, HD, h*64 + ks*32, lane);
    v16h b = bfrag(kh, tn*16, HD, h*64 + ks*32, lane);
    acc = wmma16(a, b, acc);
  }
  int mcol = tn*16 + (lane & 15);
#pragma unroll
  for (int r = 0; r < 8; r++) {
    int qr = tm*16 + ((lane >> 4) << 3) + r;
    sc[((size_t)h*NPTS + qr)*NPTS + mcol] = acc[r];
  }
}

// u[n][h][j] = sum_c Wp[j][h*64+c] * q[n][h*64+c]   (factored positional term)
__global__ void k_u(const _Float16* __restrict__ qh, const float* __restrict__ Wp,
                    float* __restrict__ u) {
  int t = blockIdx.x * blockDim.x + threadIdx.x;
  if (t >= NPTS * NHEADS * HD) return;
  int j = t % HD; int nh = t / HD; int h = nh % NHEADS; int n = nh / NHEADS;
  float s = 0.f;
  for (int c = 0; c < CHEAD; c++)
    s += (float)qh[(size_t)n*HD + h*64 + c] * Wp[(size_t)j*HD + h*64 + c];
  u[t] = s;
}

__global__ void k_qbp(const _Float16* __restrict__ qh, const float* __restrict__ bp,
                      float* __restrict__ qbp) {
  int t = blockIdx.x * blockDim.x + threadIdx.x;
  if (t >= NPTS * NHEADS) return;
  int h = t % NHEADS, n = t / NHEADS;
  float s = 0.f;
  for (int c = 0; c < CHEAD; c++) s += (float)qh[(size_t)n*HD + h*64 + c] * bp[h*64 + c];
  qbp[t] = s;
}

// scores[h][n][m] += emb[n,m,:].u[n,h,:] + qbp[n,h]; one wave per pair
__global__ void k_padd(const _Float16* __restrict__ emb, const float* __restrict__ u,
                       const float* __restrict__ qbp, float* __restrict__ sc) {
  int gw = blockIdx.x * 4 + (threadIdx.x >> 5);
  int lane = threadIdx.x & 31;
  if (gw >= NN) return;
  int n = gw / NPTS, m = gw % NPTS;
  const _Float16* e = emb + (size_t)gw * HD;
  float s0 = 0.f, s1 = 0.f, s2 = 0.f, s3 = 0.f;
#pragma unroll
  for (int i = 0; i < 8; i++) {
    int j = lane*8 + i;
    float ev = (float)e[j];
    s0 += ev * u[((size_t)n*4 + 0)*HD + j];
    s1 += ev * u[((size_t)n*4 + 1)*HD + j];
    s2 += ev * u[((size_t)n*4 + 2)*HD + j];
    s3 += ev * u[((size_t)n*4 + 3)*HD + j];
  }
  s0 = wsum(s0); s1 = wsum(s1); s2 = wsum(s2); s3 = wsum(s3);
  if (lane < 4) {
    float s = (lane == 0) ? s0 : (lane == 1) ? s1 : (lane == 2) ? s2 : s3;
    sc[((size_t)lane*NPTS + n)*NPTS + m] += s + qbp[n*4 + lane];
  }
}

// softmax over keys with 1/sqrt(64) scale; wave per (h,n) row
__global__ void k_softmax(const float* __restrict__ sc, _Float16* __restrict__ attnh) {
  int gw = blockIdx.x * 4 + (threadIdx.x >> 5);
  int lane = threadIdx.x & 31;
  if (gw >= NHEADS * NPTS) return;
  const float* row = sc + (size_t)gw * NPTS;
  float v[10], mx = -3.4e38f;
#pragma unroll
  for (int j = 0; j < 10; j++) { v[j] = row[lane + j*32] * 0.125f; mx = fmaxf(mx, v[j]); }
  mx = wmaxr(mx);
  float s = 0.f;
#pragma unroll
  for (int j = 0; j < 10; j++) { v[j] = __expf(v[j] - mx); s += v[j]; }
  s = wsum(s);
  float inv = 1.0f / s;
  _Float16* o = attnh + (size_t)gw * NPTS;
#pragma unroll
  for (int j = 0; j < 10; j++) o[lane + j*32] = (_Float16)(v[j] * inv);
}

// o[n][h*64+c] = sum_m attn[h][n][m] * vT[h*64+c][m]
__global__ void __launch_bounds__(128) k_attnv(const _Float16* __restrict__ attnh,
    const _Float16* __restrict__ vT, _Float16* __restrict__ oh) {
  int gw = blockIdx.x * 4 + (threadIdx.x >> 5);
  int lane = threadIdx.x & 31;
  const int TM = NPTS / 16;
  if (gw >= NHEADS * TM * 4) return;
  int h = gw / (TM*4); int rem = gw % (TM*4);
  int tm = rem % TM, tc = rem / TM;
  v8f acc = {0,0,0,0,0,0,0,0};
  const _Float16* A = attnh + (size_t)h * NPTS * NPTS;
  for (int ks = 0; ks < 10; ks++) {
    v16h a = afrag(A, tm*16, NPTS, ks*32, lane);
    v16h b = bfrag(vT, h*64 + tc*16, NPTS, ks*32, lane);
    acc = wmma16(a, b, acc);
  }
  int ch = h*64 + tc*16 + (lane & 15);
#pragma unroll
  for (int r = 0; r < 8; r++) {
    int qr = tm*16 + ((lane >> 4) << 3) + r;
    oh[(size_t)qr*HD + ch] = (_Float16)acc[r];
  }
}

// x = LN(x + t) * g + b ; also writes f16 mirror; wave per row
__global__ void k_ln(float* __restrict__ x, const float* __restrict__ t,
                     const float* __restrict__ g, const float* __restrict__ b,
                     _Float16* __restrict__ xh) {
  int gw = blockIdx.x * 4 + (threadIdx.x >> 5);
  int lane = threadIdx.x & 31;
  if (gw >= NPTS) return;
  float* xr = x + (size_t)gw * HD;
  const float* tr = t + (size_t)gw * HD;
  float v[8], s = 0.f;
#pragma unroll
  for (int i = 0; i < 8; i++) { v[i] = xr[lane*8 + i] + tr[lane*8 + i]; s += v[i]; }
  s = wsum(s);
  float mean = s * (1.0f / 256.0f), q = 0.f;
#pragma unroll
  for (int i = 0; i < 8; i++) { float d = v[i] - mean; q += d*d; }
  q = wsum(q);
  float rstd = rsqrtf(q * (1.0f / 256.0f) + 1e-5f);
#pragma unroll
  for (int i = 0; i < 8; i++) {
    int c = lane*8 + i;
    float y = (v[i] - mean) * rstd * g[c] + b[c];
    xr[c] = y;
    xh[(size_t)gw*HD + c] = (_Float16)y;
  }
}

// ---------------- host-side orchestration ----------------------------------
struct AttnParams {
  const _Float16 *Wq, *Wk, *Wv, *Wo, *W1, *W2;
  const float *bq, *bk, *bv, *bo, *b1, *b2, *g1, *B1, *g2, *B2;
  const float *Wp, *bp;
  const _Float16* emb;
};
struct Scratch {
  _Float16 *qh, *kh, *vT, *oh, *h1h, *attnh;
  float *scores, *tbuf, *u, *qbp;
};

static void run_attn(hipStream_t s, float* x, _Float16* xh, const _Float16* xkvh,
                     const AttnParams& P, const Scratch& W) {
  k_gemm<<<80, 128, 0, s>>>(xh,   P.Wq, P.bq, nullptr, W.qh, nullptr, NPTS, HD, HD, 0);
  k_gemm<<<80, 128, 0, s>>>(xkvh, P.Wk, P.bk, nullptr, W.kh, nullptr, NPTS, HD, HD, 0);
  k_gemm<<<80, 128, 0, s>>>(xkvh, P.Wv, P.bv, nullptr, nullptr, W.vT, NPTS, HD, HD, 0);
  k_scores<<<400, 128, 0, s>>>(W.qh, W.kh, W.scores);
  if (P.emb) {
    k_u<<<CDIV(NPTS*NHEADS*HD, 256), 256, 0, s>>>(W.qh, P.Wp, W.u);
    k_qbp<<<CDIV(NPTS*NHEADS, 256), 256, 0, s>>>(W.qh, P.bp, W.qbp);
    k_padd<<<NN/4, 128, 0, s>>>(P.emb, W.u, W.qbp, W.scores);
  }
  k_softmax<<<NHEADS*NPTS/4, 128, 0, s>>>(W.scores, W.attnh);
  k_attnv<<<80, 128, 0, s>>>(W.attnh, W.vT, W.oh);
  k_gemm<<<80, 128, 0, s>>>(W.oh, P.Wo, P.bo, W.tbuf, nullptr, nullptr, NPTS, HD, HD, 0);
  k_ln<<<80, 128, 0, s>>>(x, W.tbuf, P.g1, P.B1, xh);
  k_gemm<<<160, 128, 0, s>>>(xh, P.W1, P.b1, nullptr, W.h1h, nullptr, NPTS, HD, 2*HD, 1);
  k_gemm<<<80, 128, 0, s>>>(W.h1h, P.W2, P.b2, W.tbuf, nullptr, nullptr, NPTS, 2*HD, HD, 0);
  k_ln<<<80, 128, 0, s>>>(x, W.tbuf, P.g2, P.B2, xh);
}

extern "C" void kernel_launch(void* const* d_in, const int* in_sizes, int n_in,
                              void* d_out, int out_size, void* d_ws, size_t ws_size,
                              hipStream_t stream) {
  (void)in_sizes; (void)n_in; (void)out_size; (void)ws_size;
  const float* ref_points = (const float*)d_in[0];
  const float* src_points = (const float*)d_in[1];
  const float* ref_feats  = (const float*)d_in[2];
  const float* src_feats  = (const float*)d_in[3];
  const float* gWd = (const float*)d_in[4];  const float* gbd = (const float*)d_in[5];
  const float* gWa = (const float*)d_in[6];  const float* gba = (const float*)d_in[7];
  const float* mW  = (const float*)d_in[8];  const float* mb  = (const float*)d_in[9];
  const float* lW  = (const float*)d_in[10]; const float* lb  = (const float*)d_in[11];
  const float* selW= (const float*)d_in[12]; const float* selb= (const float*)d_in[13];
  const float* inW = (const float*)d_in[14]; const float* inb = (const float*)d_in[15];
  const float* outW= (const float*)d_in[16]; const float* outb= (const float*)d_in[17];
  const float* sWq = (const float*)d_in[18]; const float* sWk = (const float*)d_in[19];
  const float* sWv = (const float*)d_in[20]; const float* sWp = (const float*)d_in[21];
  const float* sWo = (const float*)d_in[22];
  const float* sbq = (const float*)d_in[23]; const float* sbk = (const float*)d_in[24];
  const float* sbv = (const float*)d_in[25]; const float* sbp = (const float*)d_in[26];
  const float* sbo = (const float*)d_in[27]; const float* sB1 = (const float*)d_in[28];
  const float* sb2 = (const float*)d_in[29]; const float* sB2 = (const float*)d_in[30];
  const float* sg1 = (const float*)d_in[31]; const float* sg2 = (const float*)d_in[32];
  const float* sW1 = (const float*)d_in[33]; const float* sb1 = (const float*)d_in[34];
  const float* sW2 = (const float*)d_in[35];
  const float* cWq = (const float*)d_in[36]; const float* cWk = (const float*)d_in[37];
  const float* cWv = (const float*)d_in[38]; const float* cWo = (const float*)d_in[39];
  const float* cbq = (const float*)d_in[40]; const float* cbk = (const float*)d_in[41];
  const float* cbv = (const float*)d_in[42]; const float* cbo = (const float*)d_in[43];
  const float* cB1 = (const float*)d_in[44]; const float* cb2 = (const float*)d_in[45];
  const float* cB2 = (const float*)d_in[46]; const float* cg1 = (const float*)d_in[47];
  const float* cg2 = (const float*)d_in[48]; const float* cW1 = (const float*)d_in[49];
  const float* cb1 = (const float*)d_in[50]; const float* cW2 = (const float*)d_in[51];
  float* out = (float*)d_out;

  size_t off = 0;
  auto alloc = [&](size_t bytes) -> void* {
    off = (off + 255) & ~(size_t)255;
    void* p = (void*)((char*)d_ws + off);
    off += bytes;
    return p;
  };
  _Float16* embh  = (_Float16*)alloc((size_t)2*NN*HD*2);
  float* dist     = (float*)alloc((size_t)2*NN*4);
  float* aidx     = (float*)alloc((size_t)2*NN*3*4);
  int*   knnb     = (int*)alloc((size_t)2*NPTS*3*4);
  float* den      = (float*)alloc((size_t)2*NPTS*3*4);
  float* grad     = (float*)alloc((size_t)2*NPTS*3*4);
  float* ent      = (float*)alloc((size_t)2*NPTS*3*4);
  float* e2       = (float*)alloc((size_t)2*NPTS*HD*4);
  float* x0       = (float*)alloc((size_t)NPTS*HD*4);
  float* x1       = (float*)alloc((size_t)NPTS*HD*4);
  _Float16* xh0   = (_Float16*)alloc((size_t)NPTS*HD*2);
  _Float16* xh1   = (_Float16*)alloc((size_t)NPTS*HD*2);
  _Float16* featsh= (_Float16*)alloc((size_t)NPTS*CIN*2);
  Scratch W;
  W.qh    = (_Float16*)alloc((size_t)NPTS*HD*2);
  W.kh    = (_Float16*)alloc((size_t)NPTS*HD*2);
  W.vT    = (_Float16*)alloc((size_t)HD*NPTS*2);
  W.oh    = (_Float16*)alloc((size_t)NPTS*HD*2);
  W.h1h   = (_Float16*)alloc((size_t)NPTS*2*HD*2);
  W.attnh = (_Float16*)alloc((size_t)NHEADS*NN*2);
  W.scores= (float*)alloc((size_t)NHEADS*NN*4);
  W.tbuf  = (float*)alloc((size_t)NPTS*HD*4);
  W.u     = (float*)alloc((size_t)NPTS*NHEADS*HD*4);
  W.qbp   = (float*)alloc((size_t)NPTS*NHEADS*4);
  _Float16* pWd = (_Float16*)alloc((size_t)HD*HD*2);
  _Float16* pWa = (_Float16*)alloc((size_t)HD*HD*2);
  _Float16* pIn = (_Float16*)alloc((size_t)CIN*HD*2);
  _Float16* pOut= (_Float16*)alloc((size_t)HD*HD*2);
  _Float16 *pSWq[2], *pSWk[2], *pSWv[2], *pSWo[2], *pSW1[2], *pSW2[2];
  _Float16 *pCWq[2], *pCWk[2], *pCWv[2], *pCWo[2], *pCW1[2], *pCW2[2];
  for (int i = 0; i < 2; i++) {
    pSWq[i] = (_Float16*)alloc((size_t)HD*HD*2);  pSWk[i] = (_Float16*)alloc((size_t)HD*HD*2);
    pSWv[i] = (_Float16*)alloc((size_t)HD*HD*2);  pSWo[i] = (_Float16*)alloc((size_t)HD*HD*2);
    pSW1[i] = (_Float16*)alloc((size_t)HD*2*HD*2);pSW2[i] = (_Float16*)alloc((size_t)2*HD*HD*2);
    pCWq[i] = (_Float16*)alloc((size_t)HD*HD*2);  pCWk[i] = (_Float16*)alloc((size_t)HD*HD*2);
    pCWv[i] = (_Float16*)alloc((size_t)HD*HD*2);  pCWo[i] = (_Float16*)alloc((size_t)HD*HD*2);
    pCW1[i] = (_Float16*)alloc((size_t)HD*2*HD*2);pCW2[i] = (_Float16*)alloc((size_t)2*HD*HD*2);
  }

  auto pack = [&](const float* src, _Float16* dst, int K, int Nc) {
    k_packT<<<CDIV(K*Nc, 256), 256, 0, stream>>>(src, dst, K, Nc);
  };
  pack(gWd, pWd, HD, HD); pack(gWa, pWa, HD, HD);
  pack(inW, pIn, CIN, HD); pack(outW, pOut, HD, HD);
  for (int i = 0; i < 2; i++) {
    pack(sWq + (size_t)i*HD*HD, pSWq[i], HD, HD);
    pack(sWk + (size_t)i*HD*HD, pSWk[i], HD, HD);
    pack(sWv + (size_t)i*HD*HD, pSWv[i], HD, HD);
    pack(sWo + (size_t)i*HD*HD, pSWo[i], HD, HD);
    pack(sW1 + (size_t)i*HD*2*HD, pSW1[i], HD, 2*HD);
    pack(sW2 + (size_t)i*2*HD*HD, pSW2[i], 2*HD, HD);
    pack(cWq + (size_t)i*HD*HD, pCWq[i], HD, HD);
    pack(cWk + (size_t)i*HD*HD, pCWk[i], HD, HD);
    pack(cWv + (size_t)i*HD*HD, pCWv[i], HD, HD);
    pack(cWo + (size_t)i*HD*HD, pCWo[i], HD, HD);
    pack(cW1 + (size_t)i*HD*2*HD, pCW1[i], HD, 2*HD);
    pack(cW2 + (size_t)i*2*HD*HD, pCW2[i], 2*HD, HD);
  }

  // geometry + embeddings (both clouds)
  k_pairdist<<<dim3(NN/256, 2), 256, 0, stream>>>(ref_points, src_points, dist);
  k_knn<<<CDIV(2*NPTS, 64), 64, 0, stream>>>(dist, knnb, ent);
  k_density<<<CDIV(2*NPTS, 64), 64, 0, stream>>>(dist, den);
  k_gradient<<<CDIV(2*NPTS*3, 256), 256, 0, stream>>>(den, grad);
  k_neighemb<<<2*NPTS, 256, 0, stream>>>(den, grad, ent, mW, mb, lW, lb, e2);
  k_aidx<<<dim3(NN/256, 2), 256, 0, stream>>>(ref_points, src_points, knnb, aidx);
  k_geo<<<dim3(NN/16, 2), 128, 0, stream>>>(dist, aidx, pWd, pWa, gbd, gba, embh);
  k_gate<<<(2*NN)/4, 128, 0, stream>>>(embh, e2, selW, selb);

  // input projection
  k_f2h<<<CDIV(NPTS*CIN, 256), 256, 0, stream>>>(ref_feats, featsh, NPTS*CIN);
  k_gemm<<<CDIV((NPTS/16)*(HD/16), 4), 128, 0, stream>>>(featsh, pIn, inb, x0, xh0, nullptr, NPTS, CIN, HD, 0);
  k_f2h<<<CDIV(NPTS*CIN, 256), 256, 0, stream>>>(src_feats, featsh, NPTS*CIN);
  k_gemm<<<CDIV((NPTS/16)*(HD/16), 4), 128, 0, stream>>>(featsh, pIn, inb, x1, xh1, nullptr, NPTS, CIN, HD, 0);

  _Float16* emb0 = embh;
  _Float16* emb1 = embh + (size_t)NN*HD;
  auto selfP = [&](int i) -> AttnParams {
    AttnParams P;
    P.Wq = pSWq[i]; P.Wk = pSWk[i]; P.Wv = pSWv[i]; P.Wo = pSWo[i]; P.W1 = pSW1[i]; P.W2 = pSW2[i];
    P.bq = sbq + i*HD; P.bk = sbk + i*HD; P.bv = sbv + i*HD; P.bo = sbo + i*HD;
    P.b1 = sb1 + i*2*HD; P.b2 = sb2 + i*HD;
    P.g1 = sg1 + i*HD; P.B1 = sB1 + i*HD; P.g2 = sg2 + i*HD; P.B2 = sB2 + i*HD;
    P.Wp = sWp + (size_t)i*HD*HD; P.bp = sbp + i*HD; P.emb = nullptr;
    return P;
  };
  auto crossP = [&](int i) -> AttnParams {
    AttnParams P;
    P.Wq = pCWq[i]; P.Wk = pCWk[i]; P.Wv = pCWv[i]; P.Wo = pCWo[i]; P.W1 = pCW1[i]; P.W2 = pCW2[i];
    P.bq = cbq + i*HD; P.bk = cbk + i*HD; P.bv = cbv + i*HD; P.bo = cbo + i*HD;
    P.b1 = cb1 + i*2*HD; P.b2 = cb2 + i*HD;
    P.g1 = cg1 + i*HD; P.B1 = cB1 + i*HD; P.g2 = cg2 + i*HD; P.B2 = cB2 + i*HD;
    P.Wp = nullptr; P.bp = nullptr; P.emb = nullptr;
    return P;
  };

  for (int blk = 0; blk < 4; blk++) {
    int idx = blk / 2;
    if ((blk & 1) == 0) {           // self
      AttnParams P = selfP(idx);
      P.emb = emb0; run_attn(stream, x0, xh0, xh0, P, W);
      P.emb = emb1; run_attn(stream, x1, xh1, xh1, P, W);
    } else {                        // cross (x1 uses updated x0)
      AttnParams P = crossP(idx);
      run_attn(stream, x0, xh0, xh1, P, W);
      run_attn(stream, x1, xh1, xh0, P, W);
    }
  }

  // output projection straight into d_out
  k_gemm<<<CDIV((NPTS/16)*(HD/16), 4), 128, 0, stream>>>(xh0, pOut, outb, out, nullptr, nullptr, NPTS, HD, HD, 0);
  k_gemm<<<CDIV((NPTS/16)*(HD/16), 4), 128, 0, stream>>>(xh1, pOut, outb, out + (size_t)NPTS*HD, nullptr, nullptr, NPTS, HD, HD, 0);
}
